// DiffusionLM_12773232738522
// MI455X (gfx1250) — compile-verified, matrix operand order checked by reading, not run on hardware
//
#include <hip/hip_runtime.h>
#include <hip/hip_bf16.h>
#include <math.h>

typedef _Float16 f16;
typedef _Float16 v8h  __attribute__((ext_vector_type(8)));
typedef _Float16 v16h __attribute__((ext_vector_type(16)));
typedef float    v8f  __attribute__((ext_vector_type(8)));

#define HIDDIM  512
#define HID2    1024
#define TOK     4096      // B*S = 2*2048
#define VOCABSZ 32000
#define NSTEPS  20

// ---------------- prep / elementwise kernels ----------------

__global__ void k_f32_to_f16(const float* __restrict__ src, f16* __restrict__ dst, int n) {
  int i = blockIdx.x * blockDim.x + threadIdx.x;
  if (i < n) dst[i] = (f16)src[i];
}

// W: [K][N] f32 row-major  ->  Wt: [N][K] f16
__global__ void k_transpose_f16(const float* __restrict__ W, f16* __restrict__ Wt, int K, int N) {
  int k = blockIdx.x * blockDim.x + threadIdx.x;
  int n = blockIdx.y;
  if (k < K) Wt[(size_t)n * K + k] = (f16)W[(size_t)k * N + n];
}

__global__ void k_copy_row(const float* __restrict__ W, float* __restrict__ dst, int rowIdx, int N) {
  int i = blockIdx.x * blockDim.x + threadIdx.x;
  if (i < N) dst[i] = W[(size_t)rowIdx * N + i];
}

__global__ void k_gather(const int* __restrict__ x, const float* __restrict__ embed,
                         float* __restrict__ h, f16* __restrict__ hh) {
  int row = blockIdx.x;
  int tok = x[row];
  const float* src = embed + (size_t)tok * HIDDIM;
  for (int j = threadIdx.x; j < HIDDIM; j += blockDim.x) {
    float v = src[j];
    h [(size_t)row * HIDDIM + j] = v;
    hh[(size_t)row * HIDDIM + j] = (f16)v;
  }
}

__global__ void k_update_h(float* __restrict__ h, const float* __restrict__ score,
                           f16* __restrict__ hh, float c, float isa, int n) {
  int i = blockIdx.x * blockDim.x + threadIdx.x;
  if (i < n) {
    float v = (h[i] - c * score[i]) * isa;
    h[i]  = v;
    hh[i] = (f16)v;
  }
}

// ---------------- LayerNorm (+ optional exact GELU), writes f16 ----------------
// one block (256 thr = 8 waves) per row
__global__ void k_ln_act(const float* __restrict__ X, const float* __restrict__ g,
                         const float* __restrict__ b, f16* __restrict__ Y,
                         int W, int do_gelu) {
  __shared__ float s_sum[8];
  __shared__ float s_sq[8];
  int row = blockIdx.x;
  const float* xr = X + (size_t)row * W;
  float sum = 0.f, sq = 0.f;
  for (int j = threadIdx.x; j < W; j += blockDim.x) {
    float v = xr[j]; sum += v; sq += v * v;
  }
  #pragma unroll
  for (int off = 16; off > 0; off >>= 1) {
    sum += __shfl_xor(sum, off, 32);
    sq  += __shfl_xor(sq,  off, 32);
  }
  int wave = threadIdx.x >> 5;
  if ((threadIdx.x & 31) == 0) { s_sum[wave] = sum; s_sq[wave] = sq; }
  __syncthreads();
  float tsum = 0.f, tsq = 0.f;
  #pragma unroll
  for (int i = 0; i < 8; ++i) { tsum += s_sum[i]; tsq += s_sq[i]; }
  float mu   = tsum / (float)W;
  float var  = tsq / (float)W - mu * mu;
  float rstd = rsqrtf(var + 1e-5f);
  for (int j = threadIdx.x; j < W; j += blockDim.x) {
    float v = (xr[j] - mu) * rstd * g[j] + b[j];
    if (do_gelu) v = 0.5f * v * (1.0f + erff(v * 0.70710678118654752f));
    Y[(size_t)row * W + j] = (f16)v;
  }
}

// ---------------- WMMA helpers ----------------

__device__ __forceinline__ v16h ldfrag(const f16* p) {
  // 16-bit A/B fragment: two 16B chunks (K = khalf..khalf+7 and 16+khalf..+7)
  v8h lo = *(const v8h*)p;
  v8h hi = *(const v8h*)(p + 16);
  return __builtin_shufflevector(lo, hi, 0,1,2,3,4,5,6,7,8,9,10,11,12,13,14,15);
}

__device__ __forceinline__ v8f wmma16(v16h a, v16h b, v8f c) {
  return __builtin_amdgcn_wmma_f32_16x16x32_f16(false, a, false, b,
                                                (short)0, c, false, false);
}

// ---------------- WMMA GEMM:  C[M][N] = A[M][K] * Bt[N][K]^T (+bias +scalar*extra)
// A, Bt f16 row-major; C f32. Each wave computes a 32x32 tile (2x2 WMMA tiles,
// double-buffered K pipeline). 8 waves/block arranged 2x4 -> 64x128 block tile.
// grid.x = N/128, grid.y = M/64. K multiple of 64, rows 16B-aligned.
__global__ void k_gemm_wmma(const f16* __restrict__ A, const f16* __restrict__ Bt,
                            float* __restrict__ C, int M, int N, int K,
                            const float* __restrict__ bias,
                            const float* __restrict__ extra, float extraScale) {
  const int lane = threadIdx.x & 31;
  const int wave = threadIdx.x >> 5;
  const int wx = wave & 3;
  const int wy = wave >> 2;
  const int n0 = blockIdx.x * 128 + wx * 32;
  const int m0 = blockIdx.y * 64  + wy * 32;
  if (n0 >= N || m0 >= M) return;   // wave-uniform: EXEC all-ones for WMMA

  const int r     = lane & 15;
  const int khalf = (lane >> 4) << 3;   // 0 or 8, per 16-bit A-matrix layout

  const f16* ar0 = A  + (size_t)(m0 + r) * K + khalf;
  const f16* ar1 = ar0 + (size_t)16 * K;
  const f16* br0 = Bt + (size_t)(n0 + r) * K + khalf;
  const f16* br1 = br0 + (size_t)16 * K;

  v8f c00 = {}, c01 = {}, c10 = {}, c11 = {};

  // prologue: fragments for k = 0
  v16h a0 = ldfrag(ar0), a1 = ldfrag(ar1);
  v16h b0 = ldfrag(br0), b1 = ldfrag(br1);

  for (int k0 = 0; k0 < K; k0 += 64) {
    // load fragments for k0+32 while WMMAs on k0 are in flight
    v16h a0n = ldfrag(ar0 + k0 + 32), a1n = ldfrag(ar1 + k0 + 32);
    v16h b0n = ldfrag(br0 + k0 + 32), b1n = ldfrag(br1 + k0 + 32);
    if (k0 + 128 < K) {                 // uniform branch
      __builtin_prefetch(ar0 + k0 + 128, 0, 3);
      __builtin_prefetch(ar1 + k0 + 128, 0, 3);
      __builtin_prefetch(br0 + k0 + 128, 0, 3);
      __builtin_prefetch(br1 + k0 + 128, 0, 3);
    }
    // 4 independent WMMAs on the k0 fragments
    c00 = wmma16(a0, b0, c00);
    c01 = wmma16(a0, b1, c01);
    c10 = wmma16(a1, b0, c10);
    c11 = wmma16(a1, b1, c11);
    // load fragments for k0+64 while WMMAs on k0+32 are in flight
    if (k0 + 64 < K) {                  // uniform branch
      a0 = ldfrag(ar0 + k0 + 64); a1 = ldfrag(ar1 + k0 + 64);
      b0 = ldfrag(br0 + k0 + 64); b1 = ldfrag(br1 + k0 + 64);
    }
    c00 = wmma16(a0n, b0n, c00);
    c01 = wmma16(a0n, b1n, c01);
    c10 = wmma16(a1n, b0n, c10);
    c11 = wmma16(a1n, b1n, c11);
  }

  // epilogue: C layout per tile -> VGPR i holds row rbase+i, col = lane&15 tile col
  const int col0 = n0 + (lane & 15);
  const int col1 = col0 + 16;
  const int rb   = (lane >> 4) << 3;
  float add0 = 0.0f, add1 = 0.0f;
  if (bias)  { add0 += bias[col0];  add1 += bias[col1]; }
  if (extra) { add0 += extra[col0] * extraScale; add1 += extra[col1] * extraScale; }

  #pragma unroll
  for (int i = 0; i < 8; ++i) {
    C[(size_t)(m0 + rb + i)      * N + col0] = c00[i] + add0;
    C[(size_t)(m0 + rb + i)      * N + col1] = c01[i] + add1;
    C[(size_t)(m0 + 16 + rb + i) * N + col0] = c10[i] + add0;
    C[(size_t)(m0 + 16 + rb + i) * N + col1] = c11[i] + add1;
  }
}

// ---------------- host driver ----------------

extern "C" void kernel_launch(void* const* d_in, const int* in_sizes, int n_in,
                              void* d_out, int out_size, void* d_ws, size_t ws_size,
                              hipStream_t stream) {
  const int*   x     = (const int*)  d_in[0];
  const float* embed = (const float*)d_in[1];
  const float* W1    = (const float*)d_in[2];   // [513][1024]
  const float* b1    = (const float*)d_in[3];
  const float* g1    = (const float*)d_in[4];
  const float* be1   = (const float*)d_in[5];
  const float* W2    = (const float*)d_in[6];   // [1024][1024]
  const float* b2    = (const float*)d_in[7];
  const float* g2    = (const float*)d_in[8];
  const float* be2   = (const float*)d_in[9];
  const float* W3    = (const float*)d_in[10];  // [1024][512]
  const float* b3    = (const float*)d_in[11];
  const float* gn    = (const float*)d_in[12];
  const float* bn    = (const float*)d_in[13];
  float* out = (float*)d_out;

  // workspace layout
  char* ws = (char*)d_ws;
  size_t off = 0;
  auto alloc = [&](size_t bytes) -> void* {
    void* p = ws + off;
    off = (off + bytes + 255) & ~(size_t)255;
    return p;
  };
  float* h      = (float*)alloc((size_t)TOK * HIDDIM * 4);
  f16*   hh     = (f16*)  alloc((size_t)TOK * HIDDIM * 2);
  float* z      = (float*)alloc((size_t)TOK * HID2 * 4);
  f16*   zh     = (f16*)  alloc((size_t)TOK * HID2 * 2);
  float* score  = (float*)alloc((size_t)TOK * HIDDIM * 4);
  f16*   W1t    = (f16*)  alloc((size_t)HID2 * HIDDIM * 2);
  float* w1last = (float*)alloc((size_t)HID2 * 4);
  f16*   W2t    = (f16*)  alloc((size_t)HID2 * HID2 * 2);
  f16*   W3t    = (f16*)  alloc((size_t)HIDDIM * HID2 * 2);
  f16*   embh   = (f16*)  alloc((size_t)VOCABSZ * HIDDIM * 2);

  dim3 t256(256);

  // ---- prep: f16 transposed weights, f16 embed, embedding gather ----
  k_transpose_f16<<<dim3(HIDDIM / 256, HID2), t256, 0, stream>>>(W1, W1t, HIDDIM, HID2);
  k_copy_row    <<<dim3((HID2 + 255) / 256), t256, 0, stream>>>(W1, w1last, HIDDIM, HID2);
  k_transpose_f16<<<dim3(HID2 / 256, HID2), t256, 0, stream>>>(W2, W2t, HID2, HID2);
  k_transpose_f16<<<dim3(HID2 / 256, HIDDIM), t256, 0, stream>>>(W3, W3t, HID2, HIDDIM);
  k_f32_to_f16  <<<dim3((VOCABSZ * HIDDIM + 255) / 256), t256, 0, stream>>>(embed, embh, VOCABSZ * HIDDIM);
  k_gather      <<<dim3(TOK), t256, 0, stream>>>(x, embed, h, hh);

  // ---- diffusion schedule (host-side, deterministic) ----
  float betas[NSTEPS], alphas[NSTEPS], acp[NSTEPS];
  for (int i = 0; i < NSTEPS; ++i) {
    betas[i]  = 0.0001f + (0.02f - 0.0001f) * (float)i / (float)(NSTEPS - 1);
    alphas[i] = 1.0f - betas[i];
    acp[i]    = (i ? acp[i - 1] : 1.0f) * alphas[i];
  }

  for (int s = 0; s < NSTEPS; ++s) {
    int   t   = NSTEPS - 1 - s;
    float tn  = (float)t / (float)NSTEPS;
    float c   = betas[t] / sqrtf(1.0f - acp[t]);
    float isa = 1.0f / sqrtf(alphas[t]);

    // z = hh @ W1[:512] + b1 + tn * W1[512]   [4096 x 1024], K=512
    k_gemm_wmma<<<dim3(HID2 / 128, TOK / 64), t256, 0, stream>>>(
        hh, W1t, z, TOK, HID2, HIDDIM, b1, w1last, tn);
    k_ln_act<<<dim3(TOK), t256, 0, stream>>>(z, g1, be1, zh, HID2, 1);

    // z = zh @ W2 + b2   [4096 x 1024], K=1024
    k_gemm_wmma<<<dim3(HID2 / 128, TOK / 64), t256, 0, stream>>>(
        zh, W2t, z, TOK, HID2, HID2, b2, (const float*)nullptr, 0.f);
    k_ln_act<<<dim3(TOK), t256, 0, stream>>>(z, g2, be2, zh, HID2, 1);

    // score = zh @ W3 + b3   [4096 x 512], K=1024
    k_gemm_wmma<<<dim3(HIDDIM / 128, TOK / 64), t256, 0, stream>>>(
        zh, W3t, score, TOK, HIDDIM, HID2, b3, (const float*)nullptr, 0.f);

    // h = (h - c*score) * isa   (also refresh f16 copy)
    k_update_h<<<dim3((TOK * HIDDIM + 255) / 256), t256, 0, stream>>>(
        h, score, hh, c, isa, TOK * HIDDIM);
  }

  // final LN -> f16
  k_ln_act<<<dim3(TOK), t256, 0, stream>>>(h, gn, bn, hh, HIDDIM, 0);

  // logits = hh @ embed^T   [4096 x 32000], K=512  (embh already [v][h] = Bt)
  k_gemm_wmma<<<dim3(VOCABSZ / 128, TOK / 64), t256, 0, stream>>>(
      hh, embh, out, TOK, VOCABSZ, HIDDIM, (const float*)nullptr, (const float*)nullptr, 0.f);
}
